// SelfAttention_5299989643959
// MI455X (gfx1250) — matrix operand order repaired and checked by reading, not compiled, not run on hardware
//
#include <hip/hip_runtime.h>
#include <stdint.h>

#define B_ 8
#define S_ 2048
#define D_ 1024
#define EPS_ 1e-5f
#define SCALE_ 0.03125f  // 1/sqrt(1024)

typedef __attribute__((ext_vector_type(16))) __bf16 v16bf;
typedef __attribute__((ext_vector_type(8)))  __bf16 v8bf;
typedef __attribute__((ext_vector_type(8)))  float  v8f;
typedef __attribute__((ext_vector_type(4)))  unsigned int u32x4;
typedef __attribute__((ext_vector_type(8)))  int    i32x8;
typedef __attribute__((ext_vector_type(4)))  int    i32x4;

__device__ __forceinline__ __bf16 f2bf(float f) {
  union { float f; uint32_t u; } v; v.f = f;
  uint32_t r = v.u + 0x7FFFu + ((v.u >> 16) & 1u);  // RNE
  union { uint16_t u; __bf16 b; } o; o.u = (uint16_t)(r >> 16);
  return o.b;
}

union AFrag { v16bf v; v8bf h[2]; };
union O8    { v8bf v; __bf16 e[8]; };

// A-matrix fragment (16x32 bf16) from an LDS tile with row stride `stride`
// Contiguous K order used here (same as load_b_frag): lane<16 -> row=lane, K = {0..15}; lane>=16 -> row=lane-16, K = {16..31}
__device__ __forceinline__ v16bf load_a_frag(const __bf16* tile, int stride, int k0, int lane) {
  int row = lane & 15;
  int off = (lane >> 4) * 16;
  const __bf16* p = tile + row * stride + k0 + off;
  AFrag a;
  a.h[0] = *(const v8bf*)(p);
  a.h[1] = *(const v8bf*)(p + 8);
  return a.v;
}

// B-matrix fragment (32x16 bf16). B[k][col] = mat[colbase+col][k0+k] where mat rows are
// contiguous along the reduction dim. Lane holds col = lane&15, K-half by lane>=16:
// 16 consecutive bf16 = one aligned 32B read.
__device__ __forceinline__ v16bf load_b_frag(const __bf16* mat, int stride, int colbase,
                                             int k0, int lane) {
  int c = colbase + (lane & 15);
  int k = k0 + (lane >> 4) * 16;
  return *(const v16bf*)(mat + c * stride + k);
}

// ---------------- kernel 0: fp32 -> bf16 weight conversion ----------------
__global__ __launch_bounds__(256) void convert_w_kernel(
    const float* __restrict__ a, const float* __restrict__ b, const float* __restrict__ c,
    __bf16* __restrict__ oa, __bf16* __restrict__ ob, __bf16* __restrict__ oc) {
  const float* src = (blockIdx.y == 0) ? a : ((blockIdx.y == 1) ? b : c);
  __bf16* dst      = (blockIdx.y == 0) ? oa : ((blockIdx.y == 1) ? ob : oc);
  int i = (blockIdx.x * 256 + threadIdx.x) * 4;
  float4 f = *(const float4*)(src + i);
  dst[i + 0] = f2bf(f.x); dst[i + 1] = f2bf(f.y);
  dst[i + 2] = f2bf(f.z); dst[i + 3] = f2bf(f.w);
}

// ---------------- kernel 1: QKV projection  out = x @ W^T + b ----------------
// grid.x = (B*S)/16 ; 256 threads = 8 waves ; wave w owns output cols [128w, 128w+128)
// transposed==1 stores V as Vt[b][d][s] (contiguous 16B per-lane store in D-layout).
__global__ __launch_bounds__(256) void qkv_kernel(
    const float* __restrict__ x, const __bf16* __restrict__ wbf,
    const float* __restrict__ bias, __bf16* __restrict__ out, int transposed) {
  __shared__ __bf16 xs[16 * D_];  // 32 KB
  const int tid = threadIdx.x, wave = tid >> 5, lane = tid & 31;
  const int m0 = blockIdx.x * 16;

  // stage x tile -> bf16 LDS (each thread: 16 float4 = 64 floats of one row)
  {
    int row = tid >> 4;
    int col0 = (tid & 15) * 64;
    const float4* src = (const float4*)(x + (size_t)(m0 + row) * D_ + col0);
    __bf16* dst = xs + row * D_ + col0;
#pragma unroll
    for (int i = 0; i < 16; ++i) {
      float4 f = src[i];
      dst[4 * i + 0] = f2bf(f.x); dst[4 * i + 1] = f2bf(f.y);
      dst[4 * i + 2] = f2bf(f.z); dst[4 * i + 3] = f2bf(f.w);
    }
  }
  __syncthreads();

  const int nwave = wave * 128;
#pragma unroll
  for (int j = 0; j < 8; ++j) {
    int nb = nwave + j * 16;
    v8f acc = {};
#pragma unroll 4
    for (int k0 = 0; k0 < D_; k0 += 32) {
      v16bf a = load_a_frag(xs, D_, k0, lane);
      v16bf b = load_b_frag(wbf, D_, nb, k0, lane);
      acc = __builtin_amdgcn_wmma_f32_16x16x32_bf16(false, a, false, b, (short)0, acc,
                                                    false, false);
    }
    int n = nb + (lane & 15);
    int hi = lane >> 4;
    float bv = bias[n];
    if (!transposed) {
#pragma unroll
      for (int r = 0; r < 8; ++r)
        out[(size_t)(m0 + r + 8 * hi) * D_ + n] = f2bf(acc[r] + bv);
    } else {
      int bi = m0 / S_;
      int s0 = m0 - bi * S_;
      O8 t;
#pragma unroll
      for (int r = 0; r < 8; ++r) t.e[r] = f2bf(acc[r] + bv);
      __bf16* dst = out + ((size_t)bi * D_ + n) * S_ + s0 + 8 * hi;
      *(v8bf*)dst = t.v;  // contiguous along s
    }
  }
}

// ---------------- kernel 2: fused attention + residual + LayerNorm ----------------
// grid = (S/16, B), 256 threads. LDS: [Q 32KB][scores f32 128KB][probs bf16 64KB];
// output block (f32 64KB) reuses the scores region after softmax.
__global__ __launch_bounds__(256) void attn_kernel(
    const __bf16* __restrict__ Q, const __bf16* __restrict__ K,
    const __bf16* __restrict__ Vt, const float* __restrict__ x,
    const float* __restrict__ gamma, const float* __restrict__ beta,
    float* __restrict__ outp) {
  extern __shared__ unsigned char smem[];
  __bf16* qs  = (__bf16*)smem;                        // [16][1024]
  float*  sc  = (float*)(smem + 32768);               // [16][2048]
  __bf16* pr  = (__bf16*)(smem + 32768 + 131072);     // [16][2048]
  float*  osh = (float*)(smem + 32768);               // [16][1024] (alias sc)

  const int tid = threadIdx.x, wave = tid >> 5, lane = tid & 31;
  const int q0 = blockIdx.x * 16;
  const int b  = blockIdx.y;
  const size_t rowbase = ((size_t)b * S_ + q0) * D_;

  // --- Tensor Data Mover: 1-D 32KB copy of the Q tile into LDS ---
  if (wave == 0) {
    uint64_t ga = (uint64_t)(uintptr_t)(Q + rowbase);
    uint32_t lds = (uint32_t)(uintptr_t)qs;
    u32x4 g0 = {};
    g0.x = 1u;                                         // count=1, user descriptor
    g0.y = lds;                                        // lds_addr
    g0.z = (uint32_t)ga;                               // global_addr[31:0]
    g0.w = (uint32_t)((ga >> 32) & 0x1FFFFFFu) | (2u << 30);  // addr[56:32], type=2
    i32x8 g1 = {};
    g1[0] = (1 << 16);                                 // data_size = 2 bytes
    g1[1] = (int)((16384u & 0xFFFFu) << 16);           // tensor_dim0[15:0]
    g1[2] = 0;                                         // tensor_dim0[31:16], tensor_dim1
    g1[3] = (int)(16384u << 16);                       // tile_dim0 = 16384 elems
    g1[4] = 0;                                         // tile_dim1/2 unused
    g1[5] = 16384;                                     // tensor_dim0_stride
    g1[6] = 0; g1[7] = 0;
    i32x4 z4 = {0, 0, 0, 0};
    i32x8 z8 = {};
    __builtin_amdgcn_tensor_load_to_lds(g0, g1, z4, z4, z8, 0);
    __builtin_amdgcn_s_wait_tensorcnt(0);
  }
  __syncthreads();

  // --- scores = (Q K^T) * scale ; wave w owns kv cols [256w, 256w+256) ---
  const __bf16* Kb = K + (size_t)b * S_ * D_;
  {
    int cw = wave * 256;
    for (int j = 0; j < 16; ++j) {
      int cb = cw + j * 16;
      v8f acc = {};
#pragma unroll 4
      for (int k0 = 0; k0 < D_; k0 += 32) {
        if (k0 + 128 < D_)
          __builtin_prefetch(Kb + (cb + (lane & 15)) * D_ + k0 + 128, 0, 1);
        v16bf a = load_a_frag(qs, D_, k0, lane);
        v16bf bf = load_b_frag(Kb, D_, cb, k0, lane);
        acc = __builtin_amdgcn_wmma_f32_16x16x32_bf16(false, a, false, bf, (short)0, acc,
                                                      false, false);
      }
      int n = cb + (lane & 15);
      int hi = lane >> 4;
#pragma unroll
      for (int r = 0; r < 8; ++r) sc[(r + 8 * hi) * S_ + n] = acc[r] * SCALE_;
    }
  }
  __syncthreads();

  // --- softmax: wave w handles rows 2w and 2w+1 ---
  for (int rr = 0; rr < 2; ++rr) {
    int row = wave * 2 + rr;
    float* srow = sc + row * S_;
    float mx = -1e30f;
    for (int i = lane; i < S_; i += 32) mx = fmaxf(mx, srow[i]);
    for (int o = 16; o; o >>= 1) mx = fmaxf(mx, __shfl_xor(mx, o, 32));
    float sum = 0.f;
    for (int i = lane; i < S_; i += 32) { float e = __expf(srow[i] - mx); srow[i] = e; sum += e; }
    for (int o = 16; o; o >>= 1) sum += __shfl_xor(sum, o, 32);
    float inv = 1.f / sum;
    __bf16* prow = pr + row * S_;
    for (int i = lane; i < S_; i += 32) prow[i] = f2bf(srow[i] * inv);
  }
  __syncthreads();  // probs ready; scores region dead -> reuse as osh

  // --- O = P @ V (+ residual x); wave w owns d cols [128w, 128w+128) ---
  const __bf16* Vb = Vt + (size_t)b * D_ * S_;
  const float* xrow = x + rowbase;
  {
    int gw = wave * 128;
    for (int j = 0; j < 8; ++j) {
      int gb = gw + j * 16;
      v8f acc = {};
#pragma unroll 4
      for (int c0 = 0; c0 < S_; c0 += 32) {
        v16bf a = load_a_frag(pr, S_, c0, lane);
        v16bf bf = load_b_frag(Vb, S_, gb, c0, lane);
        acc = __builtin_amdgcn_wmma_f32_16x16x32_bf16(false, a, false, bf, (short)0, acc,
                                                      false, false);
      }
      int g = gb + (lane & 15);
      int hi = lane >> 4;
#pragma unroll
      for (int r = 0; r < 8; ++r) {
        int m = r + 8 * hi;
        osh[m * D_ + g] = acc[r] + xrow[(size_t)m * D_ + g];
      }
    }
  }
  __syncthreads();

  // --- LayerNorm: wave w handles rows 2w and 2w+1 ---
  for (int rr = 0; rr < 2; ++rr) {
    int row = wave * 2 + rr;
    const float* orow = osh + row * D_;
    float s1 = 0.f, s2 = 0.f;
    for (int i = lane; i < D_; i += 32) { float v = orow[i]; s1 += v; s2 += v * v; }
    for (int o = 16; o; o >>= 1) { s1 += __shfl_xor(s1, o, 32); s2 += __shfl_xor(s2, o, 32); }
    float mu = s1 * (1.f / D_);
    float var = s2 * (1.f / D_) - mu * mu;   // population variance (matches jnp.var)
    float rs = rsqrtf(var + EPS_);
    float* dst = outp + rowbase + (size_t)row * D_;
    for (int i = lane; i < D_; i += 32) dst[i] = (orow[i] - mu) * rs * gamma[i] + beta[i];
  }
}

extern "C" void kernel_launch(void* const* d_in, const int* in_sizes, int n_in,
                              void* d_out, int out_size, void* d_ws, size_t ws_size,
                              hipStream_t stream) {
  (void)in_sizes; (void)n_in; (void)out_size; (void)ws_size;
  const float* x     = (const float*)d_in[0];
  const float* Wq    = (const float*)d_in[1];
  const float* bq    = (const float*)d_in[2];
  const float* Wk    = (const float*)d_in[3];
  const float* bk    = (const float*)d_in[4];
  const float* Wv    = (const float*)d_in[5];
  const float* bv    = (const float*)d_in[6];
  const float* gamma = (const float*)d_in[7];
  const float* beta  = (const float*)d_in[8];
  float* out = (float*)d_out;

  // workspace layout (bf16): 3 weight copies (6MB) + Q (32MB) + K (32MB) + Vt (32MB)
  __bf16* wqb = (__bf16*)d_ws;
  __bf16* wkb = wqb + (size_t)D_ * D_;
  __bf16* wvb = wkb + (size_t)D_ * D_;
  __bf16* Qb  = wvb + (size_t)D_ * D_;
  __bf16* Kb  = Qb + (size_t)B_ * S_ * D_;
  __bf16* Vtb = Kb + (size_t)B_ * S_ * D_;

  convert_w_kernel<<<dim3(D_ * D_ / 1024, 3), 256, 0, stream>>>(Wq, Wk, Wv, wqb, wkb, wvb);

  const int MT = (B_ * S_) / 16;
  qkv_kernel<<<MT, 256, 0, stream>>>(x, wqb, bq, Qb, 0);
  qkv_kernel<<<MT, 256, 0, stream>>>(x, wkb, bk, Kb, 0);
  qkv_kernel<<<MT, 256, 0, stream>>>(x, wvb, bv, Vtb, 1);

  const size_t SMEM = 32768 + 131072 + 65536;  // Q + scores + probs = 224 KB
  (void)hipFuncSetAttribute((const void*)attn_kernel,
                            hipFuncAttributeMaxDynamicSharedMemorySize, (int)SMEM);
  attn_kernel<<<dim3(S_ / 16, B_), 256, SMEM, stream>>>(Qb, Kb, Vtb, x, gamma, beta, out);
}
